// MultiHeadAtte_35665408426207
// MI455X (gfx1250) — matrix operand order repaired and checked by reading, not compiled, not run on hardware
//
#include <hip/hip_runtime.h>

// Problem constants (match reference)
#define BB 4
#define SS 2048
#define DD 1024
#define HH 16
#define KSIZ 64
#define VSIZ 64
#define OO 1024
#define HV  (HH * VSIZ)   // 1024

typedef __attribute__((ext_vector_type(16))) __bf16 v16bf;
typedef __attribute__((ext_vector_type(8)))  float  v8f;

// ---------------------------------------------------------------------------
// Fragment loaders per CDNA5 WMMA VGPR layouts (CDNA5 ISA §7.12.2), wave32.
// A 16x32 bf16: lane m/m+16 holds row m; lane<16: elems->K{0..7,16..23},
//               lane>=16: K{8..15,24..31}.
// B 32x16 bf16: lane L holds column n=L%16; lane<16: elems->K{0..7,16..23}, else K{8..15,24..31}.
// C/D 16x16 f32: lane L, vgpr r -> M = r + (L<16?0:8), N = L%16.
// ---------------------------------------------------------------------------

__device__ __forceinline__ v16bf load_a_f32(const float* base, int ld, int lane) {
    const int row  = lane & 15;
    const int koff = (lane < 16) ? 0 : 8;
    const float* p = base + (size_t)row * ld + koff;
    v16bf a;
#pragma unroll
    for (int j = 0; j < 8; ++j) a[j] = (__bf16)p[j];
#pragma unroll
    for (int j = 0; j < 8; ++j) a[8 + j] = (__bf16)p[16 + j];
    return a;
}

__device__ __forceinline__ v16bf load_a_bf16(const __bf16* base, int ld, int lane) {
    const int row  = lane & 15;
    const int koff = (lane < 16) ? 0 : 8;
    const __bf16* p = base + (size_t)row * ld + koff;
    v16bf a;
#pragma unroll
    for (int j = 0; j < 8; ++j) a[j] = p[j];
#pragma unroll
    for (int j = 0; j < 8; ++j) a[8 + j] = p[16 + j];
    return a;
}

// B[kk][n] = mem[n*ld + kk]  (N-major source: contiguous along K per lane)
__device__ __forceinline__ v16bf load_bT_bf16(const __bf16* base, int ld, int lane) {
    const int n  = lane & 15;
    const int k0 = (lane < 16) ? 0 : 8;
    const __bf16* p = base + (size_t)n * ld + k0;
    v16bf b;
#pragma unroll
    for (int j = 0; j < 8; ++j) b[j] = p[j];
#pragma unroll
    for (int j = 0; j < 8; ++j) b[8 + j] = p[16 + j];
    return b;
}

__device__ __forceinline__ v8f wmma_bf16(v16bf a, v16bf b, v8f c) {
    return __builtin_amdgcn_wmma_f32_16x16x32_bf16(false, a, false, b, (short)0, c,
                                                   false, false);
}

// Async copy engine: 16 bytes per lane, global -> LDS, tracked by ASYNCcnt.
// (CDNA5 ISA; flat shared addr[31:0] == LDS byte offset)
__device__ __forceinline__ void async_b128(__bf16* lds_dst, const __bf16* gsrc) {
    unsigned loff = (unsigned)(uintptr_t)lds_dst;
    unsigned long long ga = (unsigned long long)(uintptr_t)gsrc;
    asm volatile("global_load_async_to_lds_b128 %0, %1, off"
                 :: "v"(loff), "v"(ga)
                 : "memory");
}

// ---------------------------------------------------------------------------
// Kernel 0: one-shot weight conversion.
//   Wq/Wk/Wv [H,D,64] f32 -> bf16 N-major [H,64,D]
//   Wo [HV,O] f32        -> bf16 N-major [O,HV]
// ---------------------------------------------------------------------------
__global__ void __launch_bounds__(256)
convert_weights_kernel(const float* __restrict__ Wq, const float* __restrict__ Wk,
                       const float* __restrict__ Wv, const float* __restrict__ Wo,
                       __bf16* __restrict__ wqT, __bf16* __restrict__ wkT,
                       __bf16* __restrict__ wvT, __bf16* __restrict__ woT) {
    const size_t nW  = (size_t)HH * KSIZ * DD;          // 1M per projection weight
    const size_t nWo = (size_t)OO * HV;                 // 1M
    size_t idx = (size_t)blockIdx.x * 256 + threadIdx.x;
    if (idx < 3 * nW) {
        const int   which = (int)(idx / nW);
        const size_t r    = idx % nW;                   // [h][n][d]
        const int h = (int)(r / (KSIZ * DD));
        const int n = (int)((r / DD) % KSIZ);
        const int d = (int)(r % DD);
        const float* W = (which == 0) ? Wq : (which == 1) ? Wk : Wv;
        __bf16*      o = (which == 0) ? wqT : (which == 1) ? wkT : wvT;
        o[r] = (__bf16)W[((size_t)h * DD + d) * KSIZ + n];
    } else if (idx < 3 * nW + nWo) {
        const size_t r = idx - 3 * nW;                  // [o][c]
        const int o = (int)(r / HV);
        const int c = (int)(r % HV);
        woT[r] = (__bf16)Wo[(size_t)c * OO + o];
    }
}

// ---------------------------------------------------------------------------
// Kernel 1: per-head QKV projection.  grid=(S/16, H, B), block=32 (one wave).
// q,k -> bf16 [B,H,S,64]; v -> bf16 transposed [B,H,64,S].
// ---------------------------------------------------------------------------
__global__ void __launch_bounds__(32)
qkv_proj_kernel(const float* __restrict__ Q, const float* __restrict__ K,
                const float* __restrict__ V,
                const __bf16* __restrict__ wqT, const float* __restrict__ bq,
                const __bf16* __restrict__ wkT, const float* __restrict__ bk,
                const __bf16* __restrict__ wvT, const float* __restrict__ bv,
                __bf16* __restrict__ qws, __bf16* __restrict__ kws,
                __bf16* __restrict__ vtws) {
    const int lane = threadIdx.x;
    const int s0   = blockIdx.x * 16;
    const int h    = blockIdx.y;
    const int b    = blockIdx.z;

    const float* Qb = Q + ((size_t)b * SS + s0) * DD;
    const float* Kb = K + ((size_t)b * SS + s0) * DD;
    const float* Vb = V + ((size_t)b * SS + s0) * DD;
    const __bf16* wqh = wqT + (size_t)h * KSIZ * DD;    // [64][D]
    const __bf16* wkh = wkT + (size_t)h * KSIZ * DD;
    const __bf16* wvh = wvT + (size_t)h * VSIZ * DD;

    v8f cq[4] = {{0}}, ck[4] = {{0}}, cv[4] = {{0}};

    for (int d0 = 0; d0 < DD; d0 += 32) {
        v16bf aq = load_a_f32(Qb + d0, DD, lane);
        v16bf ak = load_a_f32(Kb + d0, DD, lane);
        v16bf av = load_a_f32(Vb + d0, DD, lane);
#pragma unroll
        for (int nb = 0; nb < 4; ++nb) {
            v16bf wq = load_bT_bf16(wqh + (size_t)(nb * 16) * DD + d0, DD, lane);
            cq[nb] = wmma_bf16(aq, wq, cq[nb]);
            v16bf wk = load_bT_bf16(wkh + (size_t)(nb * 16) * DD + d0, DD, lane);
            ck[nb] = wmma_bf16(ak, wk, ck[nb]);
            v16bf wv = load_bT_bf16(wvh + (size_t)(nb * 16) * DD + d0, DD, lane);
            cv[nb] = wmma_bf16(av, wv, cv[nb]);
        }
    }

    const int col = lane & 15;
    const int r0  = (lane < 16) ? 0 : 8;
    __bf16* qo = qws + (((size_t)b * HH + h) * SS + s0) * KSIZ;
    __bf16* ko = kws + (((size_t)b * HH + h) * SS + s0) * KSIZ;
    __bf16* vo = vtws + ((size_t)b * HH + h) * (size_t)VSIZ * SS;

#pragma unroll
    for (int nb = 0; nb < 4; ++nb) {
        const float biq = bq[(size_t)h * KSIZ + nb * 16 + col];
        const float bik = bk[(size_t)h * KSIZ + nb * 16 + col];
        const float biv = bv[(size_t)h * VSIZ + nb * 16 + col];
#pragma unroll
        for (int r = 0; r < 8; ++r) {
            const int row = r0 + r;
            qo[(size_t)row * KSIZ + nb * 16 + col] = (__bf16)(cq[nb][r] + biq);
            ko[(size_t)row * KSIZ + nb * 16 + col] = (__bf16)(ck[nb][r] + bik);
            vo[(size_t)(nb * 16 + col) * SS + s0 + row] = (__bf16)(cv[nb][r] + biv);
        }
    }
}

// ---------------------------------------------------------------------------
// Kernel 2: flash attention per (b,h,16-row tile). grid=(S/16, H, B), block=32.
// K/V tiles double-buffered in LDS via async copy engine (ASYNCcnt); online
// softmax; P converted C-layout -> A-layout through a 1KB LDS tile (DS ops are
// in-order within a wave, no barrier).  Output bf16 into cat[B,S,H*VS].
// ---------------------------------------------------------------------------
__global__ void __launch_bounds__(32)
flash_attn_kernel(const __bf16* __restrict__ qws, const __bf16* __restrict__ kws,
                  const __bf16* __restrict__ vtws, __bf16* __restrict__ cat) {
    __shared__ __bf16 ksh[2][32 * KSIZ];   // [32 rows][64]  row-major, 4KB each
    __shared__ __bf16 vsh[2][VSIZ * 32];   // [64 vcols][32] row-major, 4KB each
    __shared__ __bf16 pshr[16 * 32];

    const int lane = threadIdx.x;
    const int s0   = blockIdx.x * 16;
    const int h    = blockIdx.y;
    const int b    = blockIdx.z;

    const __bf16* qp  = qws + (((size_t)b * HH + h) * SS + s0) * KSIZ;
    const __bf16* kp  = kws + ((size_t)b * HH + h) * (size_t)SS * KSIZ;
    const __bf16* vtp = vtws + ((size_t)b * HH + h) * (size_t)VSIZ * SS;

    // stage one 32-column K/V tile into LDS buffer ib (16 async b128 ops)
    auto stage = [&](int t, int ib) {
#pragma unroll
        for (int i = 0; i < 8; ++i) {
            const unsigned e = (unsigned)(i * 32 + lane) * 8;   // bf16 elems
            // K tile is contiguous in global: kp + t*64 + e
            async_b128(&ksh[ib][e], kp + (size_t)t * KSIZ + e);
        }
#pragma unroll
        for (int i = 0; i < 8; ++i) {
            const unsigned e  = (unsigned)(i * 32 + lane) * 8;
            const unsigned vc = e >> 5;          // vcol
            const unsigned kk = e & 31;          // col within tile
            async_b128(&vsh[ib][e], vtp + (size_t)vc * SS + t + kk);
        }
    };

    // Q tile as two A fragments (K = 0..31, 32..63)
    const v16bf a0 = load_a_bf16(qp, KSIZ, lane);
    const v16bf a1 = load_a_bf16(qp + 32, KSIZ, lane);

    v8f acc[4] = {{0}};
    float m[8], l[8];
#pragma unroll
    for (int r = 0; r < 8; ++r) { m[r] = -3.0e38f; l[r] = 0.0f; }

    const float scale = 0.125f; // 1/sqrt(64)
    const int col = lane & 15;
    const int r0  = (lane < 16) ? 0 : 8;

    stage(0, 0);
    int ib = 0;
    for (int t = 0; t < SS; t += 32) {
        if (t + 32 < SS) {
            stage(t + 32, ib ^ 1);
            // 16 newly issued ops may remain outstanding; current tile is done
            asm volatile("s_wait_asynccnt 0x10" ::: "memory");
        } else {
            asm volatile("s_wait_asynccnt 0x0" ::: "memory");
        }
        const __bf16* kt = ksh[ib];
        const __bf16* vt = vsh[ib];

        // scores: two 16x16 C fragments (tile cols 0..15 and 16..31)
        v8f sf0 = {0}, sf1 = {0};
        {
            v16bf b0 = load_bT_bf16(kt, KSIZ, lane);
            v16bf b1 = load_bT_bf16(kt + 32, KSIZ, lane);
            sf0 = wmma_bf16(a0, b0, sf0);
            sf0 = wmma_bf16(a1, b1, sf0);
            v16bf b2 = load_bT_bf16(kt + 16 * KSIZ, KSIZ, lane);
            v16bf b3 = load_bT_bf16(kt + 16 * KSIZ + 32, KSIZ, lane);
            sf1 = wmma_bf16(a0, b2, sf1);
            sf1 = wmma_bf16(a1, b3, sf1);
        }

        // online softmax update (row stats live across the 16-lane halves)
        float p0[8], p1[8], cor[8];
#pragma unroll
        for (int r = 0; r < 8; ++r) {
            const float v0 = sf0[r] * scale;
            const float v1 = sf1[r] * scale;
            float mx = fmaxf(v0, v1);
#pragma unroll
            for (int d = 1; d < 16; d <<= 1) mx = fmaxf(mx, __shfl_xor(mx, d, 32));
            const float mnew = fmaxf(m[r], mx);
            p0[r] = __expf(v0 - mnew);
            p1[r] = __expf(v1 - mnew);
            float sum = p0[r] + p1[r];
#pragma unroll
            for (int d = 1; d < 16; d <<= 1) sum += __shfl_xor(sum, d, 32);
            cor[r] = __expf(m[r] - mnew);
            l[r] = l[r] * cor[r] + sum;
            m[r] = mnew;
        }
#pragma unroll
        for (int f = 0; f < 4; ++f)
#pragma unroll
            for (int r = 0; r < 8; ++r) acc[f][r] *= cor[r];

        // C-layout P -> LDS (16x32 bf16) -> A-layout fragment (in-order DS)
#pragma unroll
        for (int r = 0; r < 8; ++r) {
            pshr[(r0 + r) * 32 + col]      = (__bf16)p0[r];
            pshr[(r0 + r) * 32 + col + 16] = (__bf16)p1[r];
        }
        const v16bf pa = load_a_bf16(pshr, 32, lane);

        // acc += P(16x32) x V(32x64); V tile is [vcol][kk] in LDS
#pragma unroll
        for (int f = 0; f < 4; ++f) {
            v16bf bv = load_bT_bf16(vt + (size_t)(f * 16) * 32, 32, lane);
            acc[f] = wmma_bf16(pa, bv, acc[f]);
        }
        ib ^= 1;
    }

    // epilogue: normalize, store bf16 into cat[B,S,H*VS]
#pragma unroll
    for (int r = 0; r < 8; ++r) {
        const float inv = 1.0f / l[r];
        const int row = s0 + r0 + r;
#pragma unroll
        for (int f = 0; f < 4; ++f) {
            cat[((size_t)b * SS + row) * HV + (size_t)h * VSIZ + f * 16 + col] =
                (__bf16)(acc[f][r] * inv);
        }
    }
}

// ---------------------------------------------------------------------------
// Kernel 3: output projection out = cat @ Wo + bo.  grid=(B*S/16, O/64).
// ---------------------------------------------------------------------------
__global__ void __launch_bounds__(32)
out_proj_kernel(const __bf16* __restrict__ cat, const __bf16* __restrict__ woT,
                const float* __restrict__ bo, float* __restrict__ out) {
    const int lane = threadIdx.x;
    const int row0 = blockIdx.x * 16;
    const int o0   = blockIdx.y * 64;

    v8f c[4] = {{0}};
    for (int c0 = 0; c0 < HV; c0 += 32) {
        if (c0 + 32 < HV)
            __builtin_prefetch((const void*)(cat + (size_t)row0 * HV + c0 + 32), 0, 1);
        v16bf a = load_a_bf16(cat + (size_t)row0 * HV + c0, HV, lane);
#pragma unroll
        for (int f = 0; f < 4; ++f) {
            v16bf bW = load_bT_bf16(woT + (size_t)(o0 + f * 16) * HV + c0, HV, lane);
            c[f] = wmma_bf16(a, bW, c[f]);
        }
    }

    const int col = lane & 15;
    const int r0  = (lane < 16) ? 0 : 8;
#pragma unroll
    for (int f = 0; f < 4; ++f) {
        const float bias = bo[o0 + f * 16 + col];
#pragma unroll
        for (int r = 0; r < 8; ++r) {
            out[(size_t)(row0 + r0 + r) * OO + o0 + f * 16 + col] = c[f][r] + bias;
        }
    }
}

// ---------------------------------------------------------------------------
extern "C" void kernel_launch(void* const* d_in, const int* in_sizes, int n_in,
                              void* d_out, int out_size, void* d_ws, size_t ws_size,
                              hipStream_t stream) {
    const float* Q  = (const float*)d_in[0];
    const float* K  = (const float*)d_in[1];
    const float* V  = (const float*)d_in[2];
    const float* Wq = (const float*)d_in[3];
    const float* bq = (const float*)d_in[4];
    const float* Wk = (const float*)d_in[5];
    const float* bk = (const float*)d_in[6];
    const float* Wv = (const float*)d_in[7];
    const float* bv = (const float*)d_in[8];
    const float* Wo = (const float*)d_in[9];
    const float* bo = (const float*)d_in[10];
    float* out = (float*)d_out;

    // workspace partition (bf16)
    const size_t nAct = (size_t)BB * HH * SS * KSIZ;    // 8.39M each
    const size_t nW   = (size_t)HH * KSIZ * DD;         // 1.05M each
    __bf16* qws  = (__bf16*)d_ws;
    __bf16* kws  = qws + nAct;
    __bf16* vtws = kws + nAct;
    __bf16* cat  = vtws + nAct;
    __bf16* wqT  = cat + nAct;
    __bf16* wkT  = wqT + nW;
    __bf16* wvT  = wkT + nW;
    __bf16* woT  = wvT + nW;                            // nW elements ( O*HV )

    dim3 blk(32, 1, 1);

    const size_t nConv = 3 * nW + (size_t)OO * HV;
    convert_weights_kernel<<<dim3((unsigned)((nConv + 255) / 256)), dim3(256), 0,
                             stream>>>(Wq, Wk, Wv, Wo, wqT, wkT, wvT, woT);

    dim3 gProj(SS / 16, HH, BB);
    qkv_proj_kernel<<<gProj, blk, 0, stream>>>(Q, K, V, wqT, bq, wkT, bk, wvT, bv,
                                               qws, kws, vtws);

    dim3 gAttn(SS / 16, HH, BB);
    flash_attn_kernel<<<gAttn, blk, 0, stream>>>(qws, kws, vtws, cat);

    dim3 gOut((BB * SS) / 16, OO / 64, 1);
    out_proj_kernel<<<gOut, blk, 0, stream>>>(cat, woT, bo, out);
}
